// ResidualQuantizer_16277926051976
// MI455X (gfx1250) — matrix operand order repaired and checked by reading, not compiled, not run on hardware
//
#include <hip/hip_runtime.h>
#include <hip/hip_bf16.h>

typedef __attribute__((ext_vector_type(16))) _Float16 v16h;
typedef __attribute__((ext_vector_type(8)))  float    v8f;

// Problem constants (match reference)
constexpr int   RQ_L = 4;
constexpr int   RQ_K = 1024;
constexpr int   RQ_D = 128;
constexpr int   RQ_B = 131072;

constexpr int   ROWS_PER_WAVE = 32;                             // two 16-row M tiles
constexpr int   WAVES_PER_WG  = 2;
constexpr int   ROWS_PER_WG   = ROWS_PER_WAVE * WAVES_PER_WG;   // 64
constexpr int   LDS_STRIDE    = RQ_D + 4;                       // 132 floats, bank spread

// Workspace layout (d_ws): [f16 codebook L*K*D halfs = 1 MiB][csq_half L*K floats = 16 KiB]
constexpr size_t WS_CBH_BYTES = (size_t)RQ_L * RQ_K * RQ_D * sizeof(_Float16);

// ---------------------------------------------------------------------------
// Prep: one wave (32 lanes) per codebook row. Convert f32 -> f16 and compute
// 0.5 * ||c||^2 (argmin ||r-c||^2 == argmax (r.c - 0.5*||c||^2)).
// ---------------------------------------------------------------------------
__global__ void rq_prep_kernel(const float* __restrict__ cb,
                               _Float16* __restrict__ cbh,
                               float* __restrict__ csqh) {
    const int entry = blockIdx.x;          // 0 .. L*K-1
    const int lane  = threadIdx.x;         // 0..31
    const float4 v  = *(const float4*)(cb + (size_t)entry * RQ_D + lane * 4);

    _Float16* dst = cbh + (size_t)entry * RQ_D + lane * 4;
    dst[0] = (_Float16)v.x;
    dst[1] = (_Float16)v.y;
    dst[2] = (_Float16)v.z;
    dst[3] = (_Float16)v.w;

    float s = v.x * v.x + v.y * v.y + v.z * v.z + v.w * v.w;
    #pragma unroll
    for (int m = 16; m >= 1; m >>= 1) s += __shfl_xor(s, m, 32);
    if (lane == 0) csqh[entry] = 0.5f * s;
}

// ---------------------------------------------------------------------------
// Main: each wave owns 32 rows (two 16-row M tiles) in f32 LDS. Per level:
// score GEMM via v_wmma_f32_16x16x32_f16; each B fragment feeds TWO
// independent accumulation chains (better XDL pipelining, half the VMEM
// per WMMA). Argmax via per-lane running best + 16-lane shfl_xor reduction,
// exact f32 residual update from the original codebooks.
// ---------------------------------------------------------------------------
__global__ __launch_bounds__(WAVES_PER_WG * 32)
void rq_main_kernel(const float* __restrict__ x,
                    const float* __restrict__ cb32,
                    const _Float16* __restrict__ cbh,
                    const float* __restrict__ csqh,
                    float* __restrict__ out_codes,   // [B][L], code idx as float
                    float* __restrict__ out_q) {     // [B][D]
    __shared__ float resid[ROWS_PER_WG][LDS_STRIDE];

    const int tid   = threadIdx.x;
    const int wave  = tid >> 5;
    const int lane  = tid & 31;
    const int hi    = lane >> 4;           // half-wave (0 / 1)
    const int mrow  = lane & 15;           // N position / A-row position
    const int rowWG = wave * ROWS_PER_WAVE;
    const size_t rowBase = (size_t)blockIdx.x * ROWS_PER_WG + rowWG;

    // Load this wave's 32 rows of x into LDS (f32 residual).
    #pragma unroll
    for (int i = 0; i < ROWS_PER_WAVE; ++i) {
        const float4 v = *(const float4*)(x + (rowBase + i) * RQ_D + lane * 4);
        *(float4*)&resid[rowWG + i][lane * 4] = v;
    }
    __syncthreads();

    for (int l = 0; l < RQ_L; ++l) {
        // ---- Build A fragments for both M tiles: 16x32 f16 per chunk.
        // Contiguous k order (same as B): lo lanes K = c*32+0..15; hi lanes +16.
        v16h a0[4], a1[4];
        const float* rr0 = &resid[rowWG + mrow][0];
        const float* rr1 = &resid[rowWG + 16 + mrow][0];
        #pragma unroll
        for (int c = 0; c < 4; ++c) {
            const int b1 = c * 32 + hi * 16;
            const int b2 = c * 32 + hi * 16 + 8;
            #pragma unroll
            for (int j = 0; j < 8; ++j) {
                a0[c][j]     = (_Float16)rr0[b1 + j];
                a0[c][8 + j] = (_Float16)rr0[b2 + j];
                a1[c][j]     = (_Float16)rr1[b1 + j];
                a1[c][8 + j] = (_Float16)rr1[b2 + j];
            }
        }

        const float*    csql = csqh + l * RQ_K;
        const _Float16* cbl  = cbh + (size_t)l * RQ_K * RQ_D;

        float best0[8], best1[8];
        int   bidx0[8], bidx1[8];
        #pragma unroll
        for (int v = 0; v < 8; ++v) {
            best0[v] = -3.402823466e38f; bidx0[v] = 0;
            best1[v] = -3.402823466e38f; bidx1[v] = 0;
        }

        for (int t = 0; t < RQ_K / 16; ++t) {
            const int kk = t * 16 + mrow;   // codebook entry = column N of tile
            // B fragment 32x16 f16: lo lanes hold K=c*32+0..15, hi lanes +16.
            const _Float16* crow = cbl + (size_t)kk * RQ_D + hi * 16;
            const v16h b0 = *(const v16h*)(crow + 0);
            const v16h b1 = *(const v16h*)(crow + 32);
            const v16h b2 = *(const v16h*)(crow + 64);
            const v16h b3 = *(const v16h*)(crow + 96);

            v8f acc0 = {};
            v8f acc1 = {};
            // Two independent accumulation chains sharing the B fragments.
            acc0 = __builtin_amdgcn_wmma_f32_16x16x32_f16(false, a0[0], false, b0, (short)0, acc0, false, false);
            acc1 = __builtin_amdgcn_wmma_f32_16x16x32_f16(false, a1[0], false, b0, (short)0, acc1, false, false);
            acc0 = __builtin_amdgcn_wmma_f32_16x16x32_f16(false, a0[1], false, b1, (short)0, acc0, false, false);
            acc1 = __builtin_amdgcn_wmma_f32_16x16x32_f16(false, a1[1], false, b1, (short)0, acc1, false, false);
            acc0 = __builtin_amdgcn_wmma_f32_16x16x32_f16(false, a0[2], false, b2, (short)0, acc0, false, false);
            acc1 = __builtin_amdgcn_wmma_f32_16x16x32_f16(false, a1[2], false, b2, (short)0, acc1, false, false);
            acc0 = __builtin_amdgcn_wmma_f32_16x16x32_f16(false, a0[3], false, b3, (short)0, acc0, false, false);
            acc1 = __builtin_amdgcn_wmma_f32_16x16x32_f16(false, a1[3], false, b3, (short)0, acc1, false, false);

            const float ct = csql[kk];      // 0.5*||c||^2 for column N = mrow
            #pragma unroll
            for (int v = 0; v < 8; ++v) {
                const float s0 = acc0[v] - ct;   // tile0 row M = v + 8*hi
                const float s1 = acc1[v] - ct;   // tile1 row M = 16 + v + 8*hi
                if (s0 > best0[v]) { best0[v] = s0; bidx0[v] = kk; }
                if (s1 > best1[v]) { best1[v] = s1; bidx1[v] = kk; }
            }
        }

        // ---- Argmax across the 16 lanes of each half (N dimension),
        // tie-break on lowest index like jnp.argmax.
        #pragma unroll
        for (int v = 0; v < 8; ++v) {
            float m0 = best0[v]; int i0 = bidx0[v];
            float m1 = best1[v]; int i1 = bidx1[v];
            #pragma unroll
            for (int mask = 1; mask <= 8; mask <<= 1) {
                const float om0 = __shfl_xor(m0, mask, 32);
                const int   oi0 = __shfl_xor(i0, mask, 32);
                const float om1 = __shfl_xor(m1, mask, 32);
                const int   oi1 = __shfl_xor(i1, mask, 32);
                if (om0 > m0 || (om0 == m0 && oi0 < i0)) { m0 = om0; i0 = oi0; }
                if (om1 > m1 || (om1 == m1 && oi1 < i1)) { m1 = om1; i1 = oi1; }
            }
            best0[v] = m0; bidx0[v] = i0;
            best1[v] = m1; bidx1[v] = i1;
        }

        // ---- Emit codes. Lanes 0 and 16 write 8 rows per tile each.
        if (mrow == 0) {
            #pragma unroll
            for (int v = 0; v < 8; ++v) {
                const size_t r0 = rowBase + hi * 8 + v;
                const size_t r1 = rowBase + 16 + hi * 8 + v;
                out_codes[r0 * RQ_L + l] = (float)bidx0[v];
                out_codes[r1 * RQ_L + l] = (float)bidx1[v];
            }
        }

        // ---- Exact f32 residual update: resid[m] -= codebook[code[m]].
        #pragma unroll
        for (int m = 0; m < 32; ++m) {
            const int src = ((m >> 3) & 1) * 16;            // half holding row m
            const int v   = m & 7;
            const int code = (m < 16) ? __shfl(bidx0[v], src, 32)
                                      : __shfl(bidx1[v], src, 32);
            const float4 cv = *(const float4*)(cb32 + ((size_t)(l * RQ_K + code)) * RQ_D + lane * 4);
            float4* rp = (float4*)&resid[rowWG + m][lane * 4];
            float4 rv = *rp;
            rv.x -= cv.x; rv.y -= cv.y; rv.z -= cv.z; rv.w -= cv.w;
            *rp = rv;
        }
        __syncthreads();
    }

    // quantized = x - final_residual (exact f32)
    #pragma unroll
    for (int i = 0; i < ROWS_PER_WAVE; ++i) {
        const size_t r = rowBase + i;
        const float4 xv = *(const float4*)(x + r * RQ_D + lane * 4);
        const float4 rv = *(const float4*)&resid[rowWG + i][lane * 4];
        float4 q;
        q.x = xv.x - rv.x; q.y = xv.y - rv.y; q.z = xv.z - rv.z; q.w = xv.w - rv.w;
        *(float4*)(out_q + r * RQ_D + lane * 4) = q;
    }
}

// ---------------------------------------------------------------------------
extern "C" void kernel_launch(void* const* d_in, const int* in_sizes, int n_in,
                              void* d_out, int out_size, void* d_ws, size_t ws_size,
                              hipStream_t stream) {
    const float* x    = (const float*)d_in[0];   // [B][D] f32
    const float* cb32 = (const float*)d_in[1];   // [L][K][D] f32

    _Float16* cbh  = (_Float16*)d_ws;                                  // 1 MiB
    float*    csqh = (float*)((char*)d_ws + WS_CBH_BYTES);             // 16 KiB

    float* out_codes = (float*)d_out;                                  // [B][L]
    float* out_q     = (float*)d_out + (size_t)RQ_B * RQ_L;            // [B][D]

    // Prep: one wave per codebook row (L*K = 4096 rows).
    rq_prep_kernel<<<RQ_L * RQ_K, 32, 0, stream>>>(cb32, cbh, csqh);

    // Main: 64 rows per workgroup, 2 waves x 32 rows (two M tiles each).
    rq_main_kernel<<<RQ_B / ROWS_PER_WG, WAVES_PER_WG * 32, 0, stream>>>(
        x, cb32, cbh, csqh, out_codes, out_q);
}
